// RealMambaBlock_83416854823381
// MI455X (gfx1250) — compile-verified, matrix operand order checked by reading, not run on hardware
//
#include <hip/hip_runtime.h>
#include <hip/hip_bf16.h>

#define DIM    512
#define T_LEN  4096
#define BATCH  8
#define NSTATE 16
#define KCONV  7

typedef __attribute__((ext_vector_type(16))) _Float16 v16h;
typedef __attribute__((ext_vector_type(8)))  _Float16 v8h;
typedef __attribute__((ext_vector_type(2)))  _Float16 v2h;
typedef __attribute__((ext_vector_type(8)))  float    v8f;

__device__ __forceinline__ v16h frag_from_lds(const _Float16* rowp, int kb) {
  // 16-bit WMMA operand layout (wave32): lanes 0-15 hold K{0..7,16..23},
  // lanes 16-31 hold K{8..15,24..31}; two ds_load_b128 per fragment.
  v8h lo = *(const v8h*)(rowp + kb);
  v8h hi = *(const v8h*)(rowp + kb + 16);
  return __builtin_shufflevector(lo, hi, 0,1,2,3,4,5,6,7,8,9,10,11,12,13,14,15);
}

__device__ __forceinline__ float sigmoidf_(float x) { return 1.0f / (1.0f + __expf(-x)); }

// ---------------------------------------------------------------------------
// Kernel 1: depthwise conv over time (K=7, pad 3) + cast to f16.
// Whole x (67MB) fits in the 192MB L2, so strided re-reads are cheap.
// ---------------------------------------------------------------------------
__global__ __launch_bounds__(256) void conv_cast_kernel(const float* __restrict__ x,
                                                        const float* __restrict__ w,
                                                        _Float16* __restrict__ xl) {
  int idx = blockIdx.x * 256 + threadIdx.x;   // B*T*256 channel-pairs
  int cp = idx & 255;
  int t  = (idx >> 8) & (T_LEN - 1);
  int b  = idx >> 20;
  int c0 = cp * 2;
  const float* xb = x + ((size_t)b * T_LEN) * DIM + c0;
  float w0[KCONV], w1[KCONV];
#pragma unroll
  for (int k = 0; k < KCONV; ++k) { w0[k] = w[c0*KCONV + k]; w1[k] = w[(c0+1)*KCONV + k]; }
  float a0 = 0.f, a1 = 0.f;
#pragma unroll
  for (int k = 0; k < KCONV; ++k) {
    int tt = t + k - KCONV/2;
    if (tt >= 0 && tt < T_LEN) {
      const float2 xv = *(const float2*)(xb + (size_t)tt * DIM);
      a0 += xv.x * w0[k];
      a1 += xv.y * w1[k];
    }
  }
  v2h o; o[0] = (_Float16)a0; o[1] = (_Float16)a1;
  *(v2h*)(xl + ((size_t)b * T_LEN + t) * DIM + c0) = o;
}

// ---------------------------------------------------------------------------
// Kernel 2: cast weights to f16. wv = W_in rows 512..1535 (v then gate),
// wo = W_out. Both stored (N,K) row-major = WMMA-B friendly.
// ---------------------------------------------------------------------------
__global__ __launch_bounds__(256) void wcvt_kernel(const float* __restrict__ W_in,
                                                   const float* __restrict__ W_out,
                                                   _Float16* __restrict__ wv,
                                                   _Float16* __restrict__ wo) {
  int idx = blockIdx.x * 256 + threadIdx.x;   // 1024*512 + 512*512 total
  if (idx < 1024 * DIM) {
    wv[idx] = (_Float16)W_in[512 * DIM + idx];
  } else {
    int j = idx - 1024 * DIM;
    wo[j] = (_Float16)W_out[j];
  }
}

// ---------------------------------------------------------------------------
// Kernel 3: GEMM1 (v and gate halves together) + fused sigmoid/silu epilogue.
// Block tile: 128 rows x 64 channels; each channel tile computes BOTH the v
// weights (wv row c) and gate weights (wv row 512+c) so g*silu(v) fuses in
// registers. 8 waves = 4 (M,32) x 2 (N,32); 2x2x2 accumulators per wave.
// ---------------------------------------------------------------------------
__global__ __launch_bounds__(256) void gemm1_kernel(const _Float16* __restrict__ xl,
                                                    const _Float16* __restrict__ wv,
                                                    _Float16* __restrict__ a2,
                                                    float* __restrict__ vstate,
                                                    float* __restrict__ gstate) {
  __shared__ __align__(16) _Float16 sA [128][40];  // 80B row stride: 16B aligned, conflict-free
  __shared__ __align__(16) _Float16 sBv[ 64][40];
  __shared__ __align__(16) _Float16 sBg[ 64][40];

  const int tid  = threadIdx.x;
  const int lane = tid & 31;
  const int w    = tid >> 5;
  const int wm   = w >> 1;            // 0..3 : 32-row M slab
  const int wn   = w & 1;             // 0..1 : 32-col N slab
  const int m0   = blockIdx.x * 128;
  const int n0   = blockIdx.y * 64;
  const int lrow = lane & 15;
  const int kb   = (lane >> 4) * 8;

  v8f accV[2][2] = {};
  v8f accG[2][2] = {};

  for (int kt = 0; kt < DIM / 32; ++kt) {
    const int k0 = kt * 32;
    if (kt) __syncthreads();
    // stage A tile (128x32 f16)
#pragma unroll
    for (int i = 0; i < 2; ++i) {
      int cid = tid + i * 256;
      int row = cid >> 2, cc = cid & 3;
      *(v8h*)&sA[row][cc * 8] =
          *(const v8h*)(xl + (size_t)(m0 + row) * DIM + k0 + cc * 8);
    }
    // stage Bv / Bg tiles (64x32 f16 each)
    {
      int row = tid >> 2, cc = tid & 3;
      *(v8h*)&sBv[row][cc * 8] =
          *(const v8h*)(wv + (size_t)(n0 + row) * DIM + k0 + cc * 8);
      *(v8h*)&sBg[row][cc * 8] =
          *(const v8h*)(wv + (size_t)(512 + n0 + row) * DIM + k0 + cc * 8);
    }
    __syncthreads();

    v16h af[2], bvf[2], bgf[2];
#pragma unroll
    for (int mf = 0; mf < 2; ++mf)
      af[mf] = frag_from_lds(&sA[wm * 32 + mf * 16 + lrow][0], kb);
#pragma unroll
    for (int nf = 0; nf < 2; ++nf) {
      bvf[nf] = frag_from_lds(&sBv[wn * 32 + nf * 16 + lrow][0], kb);
      bgf[nf] = frag_from_lds(&sBg[wn * 32 + nf * 16 + lrow][0], kb);
    }
#pragma unroll
    for (int mf = 0; mf < 2; ++mf)
#pragma unroll
      for (int nf = 0; nf < 2; ++nf) {
        accV[mf][nf] = __builtin_amdgcn_wmma_f32_16x16x32_f16(
            false, af[mf], false, bvf[nf], (short)0, accV[mf][nf], false, false);
        accG[mf][nf] = __builtin_amdgcn_wmma_f32_16x16x32_f16(
            false, af[mf], false, bgf[nf], (short)0, accG[mf][nf], false, false);
      }
  }

  // epilogue: a2 = sigmoid(gate) * silu(v)  (f16); first 16 channels spill f32
  const int mtop = m0 + wm * 32 + 8 * (lane >> 4);
#pragma unroll
  for (int mf = 0; mf < 2; ++mf)
#pragma unroll
    for (int nf = 0; nf < 2; ++nf) {
      const int c = n0 + wn * 32 + nf * 16 + lrow;
#pragma unroll
      for (int r = 0; r < 8; ++r) {
        const int m = mtop + mf * 16 + r;
        float pv = accV[mf][nf][r];
        float pg = accG[mf][nf][r];
        float g  = sigmoidf_(pg);
        float sv = pv * sigmoidf_(pv);   // silu
        a2[(size_t)m * DIM + c] = (_Float16)(g * sv);
        if (c < NSTATE) {
          vstate[(size_t)m * NSTATE + c] = sv;
          gstate[(size_t)m * NSTATE + c] = g;
        }
      }
    }
}

// ---------------------------------------------------------------------------
// Kernel 4: linear recurrence on STATE=16 channels, 32-way chunk-parallel.
// h_t = A h_{t-1} + B v_t ; y_t = C h_t + D v_t. Per-chunk partials with
// h0=0, A^128 carry fix-up in LDS, second pass writes gate*y into a2[:, :16].
// ---------------------------------------------------------------------------
__global__ __launch_bounds__(512) void scan_kernel(const float* __restrict__ vstate,
                                                   const float* __restrict__ gstate,
                                                   const float* __restrict__ Ap,
                                                   const float* __restrict__ Bp,
                                                   const float* __restrict__ Cp,
                                                   const float* __restrict__ Dp,
                                                   _Float16* __restrict__ a2) {
  __shared__ float carr[32][NSTATE];
  __shared__ float h0s [32][NSTATE];
  const int tid = threadIdx.x;
  const int j   = tid >> 4;       // chunk 0..31
  const int s   = tid & 15;       // state dim
  const int b   = blockIdx.x;
  const float A  = tanhf(Ap[s]);
  const float Bc = tanhf(Bp[s]);
  const float C  = tanhf(Cp[s]);
  const float D  = tanhf(Dp[s]);
  const int CH = T_LEN / 32;      // 128
  const size_t base = (size_t)b * T_LEN;

  float h = 0.f;
  for (int t = j * CH; t < (j + 1) * CH; ++t)
    h = A * h + Bc * vstate[(base + t) * NSTATE + s];
  carr[j][s] = h;
  __syncthreads();
  if (tid < NSTATE) {
    float ap = A;
#pragma unroll
    for (int i = 0; i < 7; ++i) ap *= ap;   // A^128
    float hh = 0.f;
    for (int jj = 0; jj < 32; ++jj) {
      h0s[jj][tid] = hh;
      hh = ap * hh + carr[jj][tid];
    }
  }
  __syncthreads();
  h = h0s[j][s];
  for (int t = j * CH; t < (j + 1) * CH; ++t) {
    const size_t i = (base + t) * NSTATE + s;
    const float v = vstate[i];
    const float g = gstate[i];
    h = A * h + Bc * v;
    const float y = C * h + D * v;
    a2[(base + t) * DIM + s] = (_Float16)(g * y);
  }
}

// ---------------------------------------------------------------------------
// Kernel 5: GEMM2 (a2 x W_out^T) + residual + LayerNorm, fully fused.
// Block tile 32 (M) x 512 (full N); 8 waves = 2 (M,16) x 4 (N,128).
// Row statistics: __shfl_xor within 16-lane groups + tiny LDS combine.
// ---------------------------------------------------------------------------
__global__ __launch_bounds__(256) void gemm2_ln_kernel(const _Float16* __restrict__ a2,
                                                       const _Float16* __restrict__ wo,
                                                       const float* __restrict__ xres,
                                                       const float* __restrict__ gamma,
                                                       const float* __restrict__ beta,
                                                       float* __restrict__ out) {
  __shared__ __align__(16) _Float16 sA[ 32][40];
  __shared__ __align__(16) _Float16 sW[512][40];
  __shared__ float red1[4][32];
  __shared__ float red2[4][32];

  const int tid  = threadIdx.x;
  const int lane = tid & 31;
  const int w    = tid >> 5;
  const int wm   = w & 1;          // 0..1 : 16-row M slab
  const int wn   = w >> 1;         // 0..3 : 128-col N slab
  const int m0   = blockIdx.x * 32;
  const int lrow = lane & 15;
  const int kb   = (lane >> 4) * 8;

  v8f acc[8] = {};

  for (int kt = 0; kt < DIM / 32; ++kt) {
    const int k0 = kt * 32;
    if (kt) __syncthreads();
    if (tid < 128) {
      int row = tid >> 2, cc = tid & 3;
      *(v8h*)&sA[row][cc * 8] =
          *(const v8h*)(a2 + (size_t)(m0 + row) * DIM + k0 + cc * 8);
    }
#pragma unroll
    for (int i = 0; i < 8; ++i) {
      int cid = tid + i * 256;
      int row = cid >> 2, cc = cid & 3;
      *(v8h*)&sW[row][cc * 8] =
          *(const v8h*)(wo + (size_t)row * DIM + k0 + cc * 8);
    }
    __syncthreads();

    v16h a = frag_from_lds(&sA[wm * 16 + lrow][0], kb);
#pragma unroll
    for (int nf = 0; nf < 8; ++nf) {
      v16h bfr = frag_from_lds(&sW[wn * 128 + nf * 16 + lrow][0], kb);
      acc[nf] = __builtin_amdgcn_wmma_f32_16x16x32_f16(
          false, a, false, bfr, (short)0, acc[nf], false, false);
    }
  }

  // residual add into accumulators (z = x + out)
  const int mrow = m0 + wm * 16 + 8 * (lane >> 4);
#pragma unroll
  for (int nf = 0; nf < 8; ++nf) {
    const int n = wn * 128 + nf * 16 + lrow;
#pragma unroll
    for (int r = 0; r < 8; ++r)
      acc[nf][r] += xres[(size_t)(mrow + r) * DIM + n];
  }

  // per-row partial sums over this wave's 128 columns
  float p1[8], p2[8];
#pragma unroll
  for (int r = 0; r < 8; ++r) {
    float s1 = 0.f, s2 = 0.f;
#pragma unroll
    for (int nf = 0; nf < 8; ++nf) { float z = acc[nf][r]; s1 += z; s2 += z * z; }
#pragma unroll
    for (int msk = 1; msk <= 8; msk <<= 1) {   // reduce within 16-lane halves
      s1 += __shfl_xor(s1, msk, 32);
      s2 += __shfl_xor(s2, msk, 32);
    }
    p1[r] = s1; p2[r] = s2;
  }
  if (lrow == 0) {                              // lanes 0 and 16
#pragma unroll
    for (int r = 0; r < 8; ++r) {
      const int row = wm * 16 + r + 8 * (lane >> 4);
      red1[wn][row] = p1[r];
      red2[wn][row] = p2[r];
    }
  }
  __syncthreads();

  float gam[8], bet[8];
#pragma unroll
  for (int nf = 0; nf < 8; ++nf) {
    const int n = wn * 128 + nf * 16 + lrow;
    gam[nf] = gamma[n];
    bet[nf] = beta[n];
  }
#pragma unroll
  for (int r = 0; r < 8; ++r) {
    const int row = wm * 16 + r + 8 * (lane >> 4);
    const float s1 = red1[0][row] + red1[1][row] + red1[2][row] + red1[3][row];
    const float s2 = red2[0][row] + red2[1][row] + red2[2][row] + red2[3][row];
    const float mu   = s1 * (1.0f / DIM);
    const float var  = s2 * (1.0f / DIM) - mu * mu;
    const float rstd = rsqrtf(var + 1e-5f);
    const int m = m0 + row;
#pragma unroll
    for (int nf = 0; nf < 8; ++nf) {
      const int n = wn * 128 + nf * 16 + lrow;
      out[(size_t)m * DIM + n] = (acc[nf][r] - mu) * rstd * gam[nf] + bet[nf];
    }
  }
}

// ---------------------------------------------------------------------------
extern "C" void kernel_launch(void* const* d_in, const int* in_sizes, int n_in,
                              void* d_out, int out_size, void* d_ws, size_t ws_size,
                              hipStream_t stream) {
  const float* x     = (const float*)d_in[0];
  const float* convw = (const float*)d_in[1];
  const float* W_in  = (const float*)d_in[2];
  const float* W_out = (const float*)d_in[3];
  const float* Ap    = (const float*)d_in[4];
  const float* Bp    = (const float*)d_in[5];
  const float* Cp    = (const float*)d_in[6];
  const float* Dp    = (const float*)d_in[7];
  const float* gamma = (const float*)d_in[8];
  const float* beta  = (const float*)d_in[9];
  float* out = (float*)d_out;

  // workspace layout (bytes):
  //   xl  f16 [B,T,512]  @ 0          (33,554,432)
  //   a2  f16 [B,T,512]  @ 33,554,432 (33,554,432)
  //   wv  f16 [1024,512] @ 67,108,864 ( 1,048,576)  (v rows then gate rows)
  //   wo  f16 [512,512]  @ 68,157,440 (   524,288)
  //   vs  f32 [B,T,16]   @ 68,681,728 ( 2,097,152)
  //   gs  f32 [B,T,16]   @ 70,778,880 ( 2,097,152)
  char* ws = (char*)d_ws;
  _Float16* xl     = (_Float16*)(ws);
  _Float16* a2     = (_Float16*)(ws + (size_t)33554432);
  _Float16* wv     = (_Float16*)(ws + (size_t)67108864);
  _Float16* wo     = (_Float16*)(ws + (size_t)68157440);
  float*    vstate = (float*)   (ws + (size_t)68681728);
  float*    gstate = (float*)   (ws + (size_t)70778880);

  conv_cast_kernel<<<(BATCH * T_LEN * 256) / 256, 256, 0, stream>>>(x, convw, xl);
  wcvt_kernel<<<(1024 * DIM + 512 * DIM) / 256, 256, 0, stream>>>(W_in, W_out, wv, wo);
  gemm1_kernel<<<dim3((BATCH * T_LEN) / 128, 512 / 64), 256, 0, stream>>>(
      xl, wv, a2, vstate, gstate);
  scan_kernel<<<BATCH, 512, 0, stream>>>(vstate, gstate, Ap, Bp, Cp, Dp, a2);
  gemm2_ln_kernel<<<(BATCH * T_LEN) / 32, 256, 0, stream>>>(
      a2, wo, x, gamma, beta, out);
}
